// TransBlock_21689584845146
// MI455X (gfx1250) — matrix operand order repaired and checked by reading, not compiled, not run on hardware
//
#include <hip/hip_runtime.h>
#include <math.h>

#define BB  4
#define CCH 64
#define NN  4096
#define HIDC 256
#define EPSF 1e-5f

#if __has_builtin(__builtin_amdgcn_global_load_async_to_lds_b128) && __has_builtin(__builtin_amdgcn_s_wait_asynccnt)
#define HAS_ASYNC_LDS 1
#else
#define HAS_ASYNC_LDS 0
#endif

typedef __attribute__((ext_vector_type(16))) _Float16 v16h;
typedef __attribute__((ext_vector_type(8)))  float    v8f;
typedef __attribute__((ext_vector_type(4)))  int      v4i;

#if HAS_ASYNC_LDS
typedef __attribute__((address_space(1))) v4i gv4i;   // global int4
typedef __attribute__((address_space(3))) v4i lv4i;   // LDS int4
#endif

__device__ __forceinline__ v8f wmma16(v16h a, v16h b, v8f c) {
  return __builtin_amdgcn_wmma_f32_16x16x32_f16(false, a, false, b, (short)0, c, false, false);
}

__device__ __forceinline__ v8f zero8() {
  v8f z;
#pragma unroll
  for (int i = 0; i < 8; ++i) z[i] = 0.0f;
  return z;
}

// A fragment (16x32 f16) from row-major f32 matrix, tile origin (m0,k0).
// lane = m + 16*((k>>3)&1); p = (k>>4)*8 + (k&7)
__device__ __forceinline__ v16h loadA_f32(const float* Wm, int ld, int m0, int k0, int lane) {
  int m = lane & 15, kh = lane >> 4;
  const float* r0 = Wm + (size_t)(m0 + m) * ld + k0 + kh * 8;
  v16h a;
#pragma unroll
  for (int i = 0; i < 8; ++i) a[i] = (_Float16)r0[i];
#pragma unroll
  for (int i = 0; i < 8; ++i) a[8 + i] = (_Float16)r0[16 + i];
  return a;
}

// A fragment (16x32 f16) from row-major f16 matrix
__device__ __forceinline__ v16h loadA_f16(const _Float16* Qm, int ld, int m0, int k0, int lane) {
  int m = lane & 15, kh = lane >> 4;
  const _Float16* r0 = Qm + (size_t)(m0 + m) * ld + k0 + kh * 8;
  v16h a;
#pragma unroll
  for (int i = 0; i < 8; ++i) a[i] = r0[i];
#pragma unroll
  for (int i = 0; i < 8; ++i) a[8 + i] = r0[16 + i];
  return a;
}

// B fragment (32x16 f16): B[k][n] = T[n0+n][k0+k]  (T row-major, row = output col n)
// lane = n + 16*((k>>3)&1); p = (k>>4)*8 + (k&7)
__device__ __forceinline__ v16h loadB_f16(const _Float16* T, int ld, int n0, int k0, int lane) {
  int n = lane & 15, kh = lane >> 4;
  const _Float16* p = T + (size_t)(n0 + n) * ld + k0 + kh * 8;
  v16h b;
#pragma unroll
  for (int i = 0; i < 8; ++i) { b[i] = p[i]; b[8 + i] = p[16 + i]; }
  return b;
}

// ---------------- RoPE sin/cos tables: [N][64] f32 ----------------
__global__ void k_sincos(float* st, float* ct) {
  int idx = blockIdx.x * 256 + threadIdx.x;   // NN*64 entries
  int pos = idx >> 6, d = idx & 63;
  int y = pos >> 6, x = pos & 63;
  int dd = (d < 32) ? d : (d - 32);
  float tfrac = (float)(dd >> 1) * (1.0f / 15.0f);
  float ang = powf(10000.0f, -tfrac);
  float coord = (d < 32) ? (float)y : (float)x;
  float th = coord * ang;
  st[idx] = sinf(th);
  ct[idx] = cosf(th);
}

// ---------------- per-batch mean / rstd over C*N ----------------
__global__ void k_stats(const float* t0, const float* t1, float* stats) {
  int t = blockIdx.x >> 2, b = blockIdx.x & 3;
  const float* src = (t == 0 ? t0 : t1) + (size_t)b * (CCH * NN);
  __shared__ float ss[256], sq[256];
  float s = 0.f, q = 0.f;
  for (int i = threadIdx.x; i < CCH * NN; i += 256) { float v = src[i]; s += v; q += v * v; }
  ss[threadIdx.x] = s; sq[threadIdx.x] = q;
  __syncthreads();
  for (int stp = 128; stp > 0; stp >>= 1) {
    if (threadIdx.x < stp) { ss[threadIdx.x] += ss[threadIdx.x + stp]; sq[threadIdx.x] += sq[threadIdx.x + stp]; }
    __syncthreads();
  }
  if (threadIdx.x == 0) {
    float inv = 1.0f / (float)(CCH * NN);
    float mu = ss[0] * inv;
    float var = sq[0] * inv - mu * mu;
    stats[(t * 4 + b) * 2]     = mu;
    stats[(t * 4 + b) * 2 + 1] = rsqrtf(var + EPSF);
  }
}

// ---------------- normalize; write f16 transposed [b][n][c] and/or f32 [b][c][n] ----------------
__global__ void k_norm(const float* in, const float* g, const float* be, const float* st,
                       _Float16* outT, float* outF) {
  int idx = blockIdx.x * 256 + threadIdx.x;  // B*C*N = 2^20
  int b = idx >> 18;
  int c = (idx >> 12) & 63;
  int n = idx & 4095;
  float mu = st[b * 2], rs = st[b * 2 + 1];
  float v = (in[idx] - mu) * rs * g[c] + be[c];
  if (outT) outT[(((size_t)b << 12) + n) * 64 + c] = (_Float16)v;
  if (outF) outF[idx] = v;
}

// ---------------- depthwise 3x3 + bias + residual (kv path) -> f16 [b][n][c] ----------------
__global__ void k_dw3kv(const float* cn, const float* w, const float* bias, _Float16* outT) {
  int idx = blockIdx.x * 256 + threadIdx.x;
  int b = idx >> 18;
  int c = (idx >> 12) & 63;
  int n = idx & 4095;
  int y = n >> 6, xx = n & 63;
  const float* cb = cn + ((size_t)(b * 64 + c) << 12);
  float acc = bias[c];
#pragma unroll
  for (int ky = 0; ky < 3; ++ky)
#pragma unroll
    for (int kx = 0; kx < 3; ++kx) {
      int yy = y + ky - 1, xc = xx + kx - 1;
      if (yy >= 0 && yy < 64 && xc >= 0 && xc < 64)
        acc += cb[(yy << 6) + xc] * w[c * 9 + ky * 3 + kx];
    }
  acc += cb[n];
  outT[(((size_t)b << 12) + n) * 64 + c] = (_Float16)acc;
}

// ---------------- WMMA GEMM 64xN: q/k (RoPE -> [n][d] f16) or v (-> [d][n] f16) ----------------
__global__ void __launch_bounds__(128) k_qkv(const _Float16* actT, const float* Wm, const float* bias,
                                             const float* sin_t, const float* cos_t,
                                             _Float16* out, int rope) {
  int lane = threadIdx.x & 31;
  int wave = (blockIdx.x * 128 + threadIdx.x) >> 5;
  int b = wave >> 8;
  int n0 = (wave & 255) << 4;
  const _Float16* act = actT + ((size_t)b << 12) * 64;
  _Float16* ob = out + ((size_t)b << 12) * 64;
  v16h bf0 = loadB_f16(act, 64, n0, 0, lane);
  v16h bf1 = loadB_f16(act, 64, n0, 32, lane);
  int m = lane & 15, hh = lane >> 4;
  int npos = n0 + m;
#pragma unroll
  for (int mt = 0; mt < 4; ++mt) {
    v8f acc = zero8();
    v16h a = loadA_f32(Wm, 64, mt * 16, 0, lane);
    acc = wmma16(a, bf0, acc);
    a = loadA_f32(Wm, 64, mt * 16, 32, lane);
    acc = wmma16(a, bf1, acc);
    float vals[8];
#pragma unroll
    for (int r = 0; r < 8; ++r) vals[r] = acc[r] + bias[mt * 16 + hh * 8 + r];
    if (rope) {
#pragma unroll
      for (int r = 0; r < 8; r += 2) {
        int d = mt * 16 + hh * 8 + r;                   // even
        float s0 = sin_t[npos * 64 + d],     c0 = cos_t[npos * 64 + d];
        float s1 = sin_t[npos * 64 + d + 1], c1 = cos_t[npos * 64 + d + 1];
        float q0v = vals[r], q1v = vals[r + 1];
        ob[(size_t)npos * 64 + d]     = (_Float16)(q0v * c0 - q1v * s0);
        ob[(size_t)npos * 64 + d + 1] = (_Float16)(q1v * c1 + q0v * s1);
      }
    } else {
#pragma unroll
      for (int r = 0; r < 8; ++r) {
        int d = mt * 16 + hh * 8 + r;
        ob[((size_t)d << 12) + npos] = (_Float16)vals[r];
      }
    }
  }
}

// ---------------- async (or fallback) staging of one 32-key K/V tile into LDS ----------------
// K tile: [32 kv][64 d] f16 (4 KB).  V tile: [64 d][32 kv] f16 (4 KB). 128 threads, 16B chunks.
__device__ __forceinline__ void stage_tile(const _Float16* Kg, const _Float16* Vg,
                                           _Float16* kl, _Float16* vl, int kv0, int tid) {
#pragma unroll
  for (int j = 0; j < 2; ++j) {
    int c = tid + j * 128;
    int krow = c >> 3, koff = (c & 7) * 8;
    const _Float16* gk = Kg + (size_t)(kv0 + krow) * 64 + koff;
    _Float16* lk = kl + krow * 64 + koff;
    int vd = c >> 2, voff = (c & 3) * 8;
    const _Float16* gv = Vg + (size_t)vd * NN + kv0 + voff;
    _Float16* lv = vl + vd * 32 + voff;
#if HAS_ASYNC_LDS
    __builtin_amdgcn_global_load_async_to_lds_b128((gv4i*)gk, (lv4i*)lk, 0, 0);
    __builtin_amdgcn_global_load_async_to_lds_b128((gv4i*)gv, (lv4i*)lv, 0, 0);
#else
    *(float4*)lk = *(const float4*)gk;
    *(float4*)lv = *(const float4*)gv;
#endif
  }
}

// ---- flash attention: 4 waves/block, 32 q-rows/wave, K/V tiles double-buffered in LDS ----
__global__ void __launch_bounds__(128) k_attn(const _Float16* qr, const _Float16* kr,
                                              const _Float16* vt, float* o) {
  __shared__ __align__(16) _Float16 kst[2][32 * 64];   // [buf][kv][d]
  __shared__ __align__(16) _Float16 vst[2][64 * 32];   // [buf][d][kv]
  __shared__ __align__(16) _Float16 pst[4][16 * 32];   // per-wave P stage
  int tid = threadIdx.x;
  int lw = tid >> 5, lane = tid & 31;
  int b = blockIdx.x >> 5;
  int qblk = (blockIdx.x & 31) << 7;                   // 128 q-rows per block
  int qw0 = qblk + lw * 32;                            // 32 q-rows per wave
  const _Float16* Q = qr + ((size_t)b << 12) * 64;     // [n][d]
  const _Float16* K = kr + ((size_t)b << 12) * 64;     // [n][d]
  const _Float16* V = vt + ((size_t)b << 12) * 64;     // [d][n]
  _Float16* ps = pst[lw];
  int m = lane & 15, hh = lane >> 4;

  v16h a00 = loadA_f16(Q, 64, qw0,      0,  lane);
  v16h a01 = loadA_f16(Q, 64, qw0,      32, lane);
  v16h a10 = loadA_f16(Q, 64, qw0 + 16, 0,  lane);
  v16h a11 = loadA_f16(Q, 64, qw0 + 16, 32, lane);

  float mrun[2][8], lrun[2][8];
  v8f acc[2][4];
#pragma unroll
  for (int qt = 0; qt < 2; ++qt) {
#pragma unroll
    for (int r = 0; r < 8; ++r) { mrun[qt][r] = -3e30f; lrun[qt][r] = 0.f; }
#pragma unroll
    for (int t = 0; t < 4; ++t) acc[qt][t] = zero8();
  }

  stage_tile(K, V, kst[0], vst[0], 0, tid);
  for (int it = 0; it < NN / 32; ++it) {
    int cur = it & 1;
#if HAS_ASYNC_LDS
    __builtin_amdgcn_s_wait_asynccnt(0);
#endif
    __syncthreads();
    if (it + 1 < NN / 32) stage_tile(K, V, kst[1 - cur], vst[1 - cur], (it + 1) * 32, tid);
    const _Float16* kl = kst[cur];
    const _Float16* vl = vst[cur];
#pragma unroll
    for (int qt = 0; qt < 2; ++qt) {
      v16h aa0 = qt ? a10 : a00;
      v16h aa1 = qt ? a11 : a01;
      v8f s0 = zero8(), s1 = zero8();
      v16h bk = loadB_f16(kl, 64, 0, 0, lane);   s0 = wmma16(aa0, bk, s0);
      bk = loadB_f16(kl, 64, 0, 32, lane);       s0 = wmma16(aa1, bk, s0);
      bk = loadB_f16(kl, 64, 16, 0, lane);       s1 = wmma16(aa0, bk, s1);
      bk = loadB_f16(kl, 64, 16, 32, lane);      s1 = wmma16(aa1, bk, s1);
      float al[8];
#pragma unroll
      for (int r = 0; r < 8; ++r) {
        float x0 = s0[r] * 0.125f, x1 = s1[r] * 0.125f;   // HD^-0.5
        float mx = fmaxf(x0, x1);
#pragma unroll
        for (int off = 1; off < 16; off <<= 1) mx = fmaxf(mx, __shfl_xor(mx, off, 32));
        float nm = fmaxf(mrun[qt][r], mx);
        al[r] = __expf(mrun[qt][r] - nm);
        mrun[qt][r] = nm;
        float p0 = __expf(x0 - nm), p1 = __expf(x1 - nm);
        s0[r] = p0; s1[r] = p1;
        float sm = p0 + p1;
#pragma unroll
        for (int off = 1; off < 16; off <<= 1) sm += __shfl_xor(sm, off, 32);
        lrun[qt][r] = lrun[qt][r] * al[r] + sm;
      }
#pragma unroll
      for (int r = 0; r < 8; ++r) {
        acc[qt][0][r] *= al[r]; acc[qt][1][r] *= al[r];
        acc[qt][2][r] *= al[r]; acc[qt][3][r] *= al[r];
        int row = hh * 8 + r;                    // C layout -> A layout via LDS
        ps[row * 32 + m]      = (_Float16)s0[r];
        ps[row * 32 + 16 + m] = (_Float16)s1[r];
      }
      v16h pa;
      {
        const _Float16* rp = ps + m * 32 + hh * 8;
#pragma unroll
        for (int i = 0; i < 8; ++i) { pa[i] = rp[i]; pa[8 + i] = rp[16 + i]; }
      }
      v16h bv = loadB_f16(vl, 32, 0,  0, lane);  acc[qt][0] = wmma16(pa, bv, acc[qt][0]);
      bv = loadB_f16(vl, 32, 16, 0, lane);       acc[qt][1] = wmma16(pa, bv, acc[qt][1]);
      bv = loadB_f16(vl, 32, 32, 0, lane);       acc[qt][2] = wmma16(pa, bv, acc[qt][2]);
      bv = loadB_f16(vl, 32, 48, 0, lane);       acc[qt][3] = wmma16(pa, bv, acc[qt][3]);
    }
  }
  float* ob = o + ((size_t)b << 12) * 64;        // [d][n]
#pragma unroll
  for (int qt = 0; qt < 2; ++qt) {
#pragma unroll
    for (int r = 0; r < 8; ++r) {
      float inv = 1.0f / lrun[qt][r];
      int qq = qw0 + qt * 16 + hh * 8 + r;
      ob[((size_t)(m)      << 12) + qq] = acc[qt][0][r] * inv;
      ob[((size_t)(16 + m) << 12) + qq] = acc[qt][1][r] * inv;
      ob[((size_t)(32 + m) << 12) + qq] = acc[qt][2][r] * inv;
      ob[((size_t)(48 + m) << 12) + qq] = acc[qt][3][r] * inv;
    }
  }
}

__global__ void k_add(const float* a, const float* b, float* o) {
  int i = blockIdx.x * 256 + threadIdx.x;
  o[i] = a[i] + b[i];
}

// ---------------- fc1 (256x64) + BN + exact GELU -> f32 [b][256][n] ----------------
__global__ void __launch_bounds__(128) k_fc1(const _Float16* actT, const float* Wm,
                                             const float* g, const float* be, float* hout) {
  int lane = threadIdx.x & 31;
  int wave = (blockIdx.x * 128 + threadIdx.x) >> 5;
  int b = wave >> 8;
  int n0 = (wave & 255) << 4;
  const _Float16* act = actT + ((size_t)b << 12) * 64;
  v16h bf0 = loadB_f16(act, 64, n0, 0, lane);
  v16h bf1 = loadB_f16(act, 64, n0, 32, lane);
  int m = lane & 15, hh = lane >> 4;
  int npos = n0 + m;
  float bs = rsqrtf(1.0f + EPSF);
  for (int mt = 0; mt < 16; ++mt) {
    v8f acc = zero8();
    v16h a = loadA_f32(Wm, 64, mt * 16, 0, lane);
    acc = wmma16(a, bf0, acc);
    a = loadA_f32(Wm, 64, mt * 16, 32, lane);
    acc = wmma16(a, bf1, acc);
#pragma unroll
    for (int r = 0; r < 8; ++r) {
      int oc = mt * 16 + hh * 8 + r;
      float xv = acc[r] * (g[oc] * bs) + be[oc];
      float ge = 0.5f * xv * (1.0f + erff(xv * 0.70710678f));
      hout[((size_t)(b * 256 + oc) << 12) + npos] = ge;
    }
  }
}

// ---------------- dw(1/3/5/7) -> *fft_w (FFT identity) -> +res -> GELU -> BN -> f16 [b][n][256] ----
__global__ void k_dwmix(const float* hin,
                        const float* w1, const float* b1,
                        const float* w3, const float* b3,
                        const float* w5, const float* b5,
                        const float* w7, const float* b7,
                        const float* fftw, const float* bnmg, const float* bnmb,
                        _Float16* outT) {
  int idx = blockIdx.x * 256 + threadIdx.x;   // B*256*N = 2^22
  int b = idx >> 20;
  int ch = (idx >> 12) & 255;
  int n = idx & 4095;
  int y = n >> 6, xx = n & 63;
  const float* hb = hin + ((size_t)(b * 256 + ch) << 12);
  float acc;
  if (ch < 64) {
    acc = hb[n] * w1[ch] + b1[ch];
  } else if (ch < 128) {
    int c = ch - 64; acc = b3[c];
    for (int ky = 0; ky < 3; ++ky)
      for (int kx = 0; kx < 3; ++kx) {
        int yy = y + ky - 1, xc = xx + kx - 1;
        if (yy >= 0 && yy < 64 && xc >= 0 && xc < 64)
          acc += hb[(yy << 6) + xc] * w3[c * 9 + ky * 3 + kx];
      }
  } else if (ch < 192) {
    int c = ch - 128; acc = b5[c];
    for (int ky = 0; ky < 5; ++ky)
      for (int kx = 0; kx < 5; ++kx) {
        int yy = y + ky - 2, xc = xx + kx - 2;
        if (yy >= 0 && yy < 64 && xc >= 0 && xc < 64)
          acc += hb[(yy << 6) + xc] * w5[c * 25 + ky * 5 + kx];
      }
  } else {
    int c = ch - 192; acc = b7[c];
    for (int ky = 0; ky < 7; ++ky)
      for (int kx = 0; kx < 7; ++kx) {
        int yy = y + ky - 3, xc = xx + kx - 3;
        if (yy >= 0 && yy < 64 && xc >= 0 && xc < 64)
          acc += hb[(yy << 6) + xc] * w7[c * 49 + ky * 7 + kx];
      }
  }
  acc *= fftw[ch];             // irfft2(fft_w * rfft2(x)) == fft_w * x
  acc += hb[n];                // + res
  float ge = 0.5f * acc * (1.0f + erff(acc * 0.70710678f));
  float o = ge * (bnmg[ch] * rsqrtf(1.0f + EPSF)) + bnmb[ch];
  outT[(((size_t)b << 12) + n) * 256 + ch] = (_Float16)o;
}

// ---------------- fc2 (64x256) + BN + final residual -> d_out ----------------
__global__ void __launch_bounds__(128) k_fc2(const _Float16* hT, const float* Wm,
                                             const float* g, const float* be,
                                             const float* x2, float* outp) {
  int lane = threadIdx.x & 31;
  int wave = (blockIdx.x * 128 + threadIdx.x) >> 5;
  int b = wave >> 8;
  int n0 = (wave & 255) << 4;
  const _Float16* act = hT + ((size_t)b << 12) * 256;
  v8f acc0 = zero8(), acc1 = zero8(), acc2 = zero8(), acc3 = zero8();
  for (int kc = 0; kc < 256; kc += 32) {
    v16h bf = loadB_f16(act, 256, n0, kc, lane);
    v16h a = loadA_f32(Wm, 256, 0, kc, lane);   acc0 = wmma16(a, bf, acc0);
    a = loadA_f32(Wm, 256, 16, kc, lane);       acc1 = wmma16(a, bf, acc1);
    a = loadA_f32(Wm, 256, 32, kc, lane);       acc2 = wmma16(a, bf, acc2);
    a = loadA_f32(Wm, 256, 48, kc, lane);       acc3 = wmma16(a, bf, acc3);
  }
  int m = lane & 15, hh = lane >> 4;
  int npos = n0 + m;
  float bs = rsqrtf(1.0f + EPSF);
#pragma unroll
  for (int r = 0; r < 8; ++r) {
    int oc0 = hh * 8 + r;
    {
      int oc = oc0; size_t ix = ((size_t)(b * 64 + oc) << 12) + npos;
      outp[ix] = x2[ix] + acc0[r] * (g[oc] * bs) + be[oc];
    }
    {
      int oc = 16 + oc0; size_t ix = ((size_t)(b * 64 + oc) << 12) + npos;
      outp[ix] = x2[ix] + acc1[r] * (g[oc] * bs) + be[oc];
    }
    {
      int oc = 32 + oc0; size_t ix = ((size_t)(b * 64 + oc) << 12) + npos;
      outp[ix] = x2[ix] + acc2[r] * (g[oc] * bs) + be[oc];
    }
    {
      int oc = 48 + oc0; size_t ix = ((size_t)(b * 64 + oc) << 12) + npos;
      outp[ix] = x2[ix] + acc3[r] * (g[oc] * bs) + be[oc];
    }
  }
}

extern "C" void kernel_launch(void* const* d_in, const int* in_sizes, int n_in,
                              void* d_out, int out_size, void* d_ws, size_t ws_size,
                              hipStream_t stream) {
  const float* x     = (const float*)d_in[0];
  const float* cnnf  = (const float*)d_in[1];
  const float* n1_g  = (const float*)d_in[2];
  const float* n1_b  = (const float*)d_in[3];
  const float* n2_g  = (const float*)d_in[4];
  const float* n2_b  = (const float*)d_in[5];
  const float* n3_g  = (const float*)d_in[6];
  const float* n3_b  = (const float*)d_in[7];
  const float* q_w   = (const float*)d_in[8];
  const float* q_b   = (const float*)d_in[9];
  const float* kv_w  = (const float*)d_in[10];
  const float* kv_b  = (const float*)d_in[11];
  const float* lc_w  = (const float*)d_in[12];
  const float* lc_b  = (const float*)d_in[13];
  const float* fc1_w = (const float*)d_in[14];
  const float* bn1_g = (const float*)d_in[15];
  const float* bn1_b = (const float*)d_in[16];
  const float* dw_w1 = (const float*)d_in[17];
  const float* dw_b1 = (const float*)d_in[18];
  const float* dw_w3 = (const float*)d_in[19];
  const float* dw_b3 = (const float*)d_in[20];
  const float* dw_w5 = (const float*)d_in[21];
  const float* dw_b5 = (const float*)d_in[22];
  const float* dw_w7 = (const float*)d_in[23];
  const float* dw_b7 = (const float*)d_in[24];
  const float* bnm_g = (const float*)d_in[25];
  const float* bnm_b = (const float*)d_in[26];
  const float* fft_w = (const float*)d_in[27];
  const float* fc2_w = (const float*)d_in[28];
  const float* bn2_g = (const float*)d_in[29];
  const float* bn2_b = (const float*)d_in[30];
  (void)in_sizes; (void)n_in; (void)out_size; (void)ws_size;

  char* wsb = (char*)d_ws;
  size_t off = 0;
  auto alloc = [&](size_t bytes) -> void* {
    void* p = (void*)(wsb + off);
    off += (bytes + 255) & ~(size_t)255;
    return p;
  };
  float*    sin_t = (float*)alloc((size_t)NN * 64 * 4);
  float*    cos_t = (float*)alloc((size_t)NN * 64 * 4);
  float*    stats = (float*)alloc(256);
  _Float16* xnT   = (_Float16*)alloc((size_t)BB * NN * 64 * 2);
  float*    cn    = (float*)alloc((size_t)BB * 64 * NN * 4);
  _Float16* kvT   = (_Float16*)alloc((size_t)BB * NN * 64 * 2);
  _Float16* qrT   = (_Float16*)alloc((size_t)BB * NN * 64 * 2);
  _Float16* krT   = (_Float16*)alloc((size_t)BB * NN * 64 * 2);
  _Float16* vT    = (_Float16*)alloc((size_t)BB * NN * 64 * 2);
  float*    obuf  = (float*)alloc((size_t)BB * 64 * NN * 4);
  float*    x2    = (float*)alloc((size_t)BB * 64 * NN * 4);
  _Float16* xn2T  = (_Float16*)alloc((size_t)BB * NN * 64 * 2);
  float*    hbuf  = (float*)alloc((size_t)BB * HIDC * NN * 4);
  _Float16* hT    = (_Float16*)alloc((size_t)BB * NN * HIDC * 2);

  const int ELEM = BB * 64 * NN;  // 2^20

  k_sincos<<<(NN * 64) / 256, 256, 0, stream>>>(sin_t, cos_t);
  k_stats<<<8, 256, 0, stream>>>(x, cnnf, stats);
  k_norm<<<ELEM / 256, 256, 0, stream>>>(x, n1_g, n1_b, stats, xnT, nullptr);
  k_norm<<<ELEM / 256, 256, 0, stream>>>(cnnf, n3_g, n3_b, stats + 8, nullptr, cn);
  k_dw3kv<<<ELEM / 256, 256, 0, stream>>>(cn, lc_w, lc_b, kvT);
  k_qkv<<<256, 128, 0, stream>>>(xnT, q_w, q_b, sin_t, cos_t, qrT, 1);
  k_qkv<<<256, 128, 0, stream>>>(kvT, kv_w, kv_b, sin_t, cos_t, krT, 1);
  k_qkv<<<256, 128, 0, stream>>>(kvT, kv_w + 64 * 64, kv_b + 64, sin_t, cos_t, vT, 0);
  k_attn<<<BB * 32, 128, 0, stream>>>(qrT, krT, vT, obuf);
  k_add<<<ELEM / 256, 256, 0, stream>>>(x, obuf, x2);
  k_stats<<<4, 256, 0, stream>>>(x2, x2, stats + 16);
  k_norm<<<ELEM / 256, 256, 0, stream>>>(x2, n2_g, n2_b, stats + 16, xn2T, nullptr);
  k_fc1<<<256, 128, 0, stream>>>(xn2T, fc1_w, bn1_g, bn1_b, hbuf);
  k_dwmix<<<(BB * HIDC * NN) / 256, 256, 0, stream>>>(hbuf, dw_w1, dw_b1, dw_w3, dw_b3,
                                                      dw_w5, dw_b5, dw_w7, dw_b7,
                                                      fft_w, bnm_g, bnm_b, hT);
  k_fc2<<<256, 128, 0, stream>>>(hT, fc2_w, bn2_g, bn2_b, x2, (float*)d_out);
}
